// SparseLinearAttention_26645977104612
// MI455X (gfx1250) — matrix operand order repaired and checked by reading, not compiled, not run on hardware
//
#include <hip/hip_runtime.h>
#include <hip/hip_bf16.h>

typedef _Float16 h16v __attribute__((ext_vector_type(16)));
typedef _Float16 h8v  __attribute__((ext_vector_type(8)));
typedef float    f8v  __attribute__((ext_vector_type(8)));
typedef int      v4i  __attribute__((ext_vector_type(4)));

#define BLK 64
#define NB  64
#define TOPK 16
#define DH  64
#define LDSW 72   // padded row stride (f16 elems) to stagger LDS banks

#if __has_builtin(__builtin_amdgcn_global_load_async_to_lds_b128)
#define HAS_ASYNC_LDS 1
#else
#define HAS_ASYNC_LDS 0
#endif

__device__ __forceinline__ f8v wmma_f16(h16v a, h16v b, f8v c) {
    return __builtin_amdgcn_wmma_f32_16x16x32_f16(false, a, false, b, (short)0, c, false, false);
}

// async 16-byte global->LDS copy (per lane); falls back to a synchronous copy
__device__ __forceinline__ void async_cp16(const _Float16* g, _Float16* l) {
#if HAS_ASYNC_LDS
    __builtin_amdgcn_global_load_async_to_lds_b128(
        (__attribute__((address_space(1))) v4i*)(unsigned long long)(const void*)g,
        (__attribute__((address_space(3))) v4i*)l, 0, 0);
#else
    *(h8v*)l = *(const h8v*)g;
#endif
}
__device__ __forceinline__ void async_drain() {
#if HAS_ASYNC_LDS
#if __has_builtin(__builtin_amdgcn_s_wait_asynccnt)
    __builtin_amdgcn_s_wait_asynccnt(0);
#else
    asm volatile("s_wait_asynccnt 0x0" ::: "memory");
#endif
#endif
}

// ---- DPP16 reductions over each 16-lane half (rows live per half in WMMA C layout) ----
template <int CTRL>
__device__ __forceinline__ float dppmov(float x) {
    return __int_as_float(
        __builtin_amdgcn_update_dpp(0, __float_as_int(x), CTRL, 0xF, 0xF, true));
}
__device__ __forceinline__ float redmax16(float x) {
    x = fmaxf(x, dppmov<0xB1>(x));    // quad_perm(1,0,3,2)
    x = fmaxf(x, dppmov<0x4E>(x));    // quad_perm(2,3,0,1)
    x = fmaxf(x, dppmov<0x141>(x));   // row_half_mirror
    x = fmaxf(x, dppmov<0x140>(x));   // row_mirror
    return x;
}
__device__ __forceinline__ float redsum16(float x) {
    x += dppmov<0xB1>(x);
    x += dppmov<0x4E>(x);
    x += dppmov<0x141>(x);
    x += dppmov<0x140>(x);
    return x;
}

// A-matrix fragment (16x32, lane = row): halves 0..7 = K[base..+7], 8..15 = K[base+16..+23]
__device__ __forceinline__ h16v frag_a(const _Float16* p) {
    h8v a = *(const h8v*)(p);
    h8v b = *(const h8v*)(p + 16);
    h16v r;
#pragma unroll
    for (int i = 0; i < 8; ++i) { r[i] = a[i]; r[i + 8] = b[i]; }
    return r;
}
// B-matrix fragment (32x16, lane = col): halves 0..7 = K[base..+7], 8..15 = K[base+16..+23]
__device__ __forceinline__ h16v frag_b(const _Float16* p) {
    h8v a = *(const h8v*)(p);
    h8v b = *(const h8v*)(p + 16);
    h16v r;
#pragma unroll
    for (int i = 0; i < 8; ++i) { r[i] = a[i]; r[i + 8] = b[i]; }
    return r;
}

// ---------------- Kernel 0a: fp32 -> f16 for Q (scaled) and K ----------------
__global__ void cvt_kernel(const float* __restrict__ q, const float* __restrict__ k,
                           _Float16* __restrict__ qh, _Float16* __restrict__ kh, int n) {
    int i = blockIdx.x * blockDim.x + threadIdx.x;
    if (i < n) {
        qh[i] = (_Float16)(q[i] * 0.125f);   // qk_scale = D^-0.5
        kh[i] = (_Float16)k[i];
    }
}

// ---------------- Kernel 0b: V -> f16 V^T (per head: [d][l]) ----------------
__global__ void vtrans_kernel(const float* __restrict__ v, _Float16* __restrict__ vt) {
    __shared__ _Float16 tile[BLK * 65];
    int bid = blockIdx.x;
    int n   = bid & (NB - 1);
    int bh  = bid >> 6;
    int tid = threadIdx.x;
    size_t src = ((size_t)bh * 4096 + (size_t)n * BLK) * DH;
#pragma unroll
    for (int i = 0; i < 16; ++i) {
        int e = tid + i * 256;
        int key = e >> 6, d = e & 63;
        tile[key * 65 + d] = (_Float16)v[src + e];
    }
    __syncthreads();
#pragma unroll
    for (int i = 0; i < 2; ++i) {
        int e = tid + i * 256;
        int d = e >> 3, k8 = (e & 7) * 8;
        h8v o;
#pragma unroll
        for (int x = 0; x < 8; ++x) o[x] = tile[(k8 + x) * 65 + d];
        *(h8v*)&vt[((size_t)bh * DH + d) * 4096 + (size_t)n * BLK + k8] = o;
    }
}

// ---------------- Kernel 1: block mean pooling ----------------
__global__ void pool_kernel(const float* __restrict__ q, const float* __restrict__ k,
                            float* __restrict__ pq, float* __restrict__ pkr) {
    int bid = blockIdx.x;
    int n   = bid & (NB - 1);
    int bh  = bid >> 6;
    int d   = threadIdx.x;
    size_t base = ((size_t)bh * 4096 + (size_t)n * BLK) * DH + d;
    float aq = 0.f, ak = 0.f;
    for (int r = 0; r < BLK; ++r) {
        aq += q[base + (size_t)r * DH];
        ak += k[base + (size_t)r * DH];
    }
    pq [(size_t)bid * DH + d] = aq * (1.0f / BLK);
    pkr[(size_t)bid * DH + d] = ak * (1.0f / BLK);
}

// ---------------- Kernel 2: block scores + top-16 ----------------
__global__ void topk_kernel(const float* __restrict__ pq, const float* __restrict__ pkr,
                            int* __restrict__ lut) {
    __shared__ float spq[NB * DH];
    __shared__ float spk[NB * DH];
    __shared__ float km[DH];
    __shared__ float ssc[NB * NB];
    int bh = blockIdx.x;
    int t  = threadIdx.x;
    size_t base = (size_t)bh * NB * DH;
    for (int n = 0; n < NB; ++n) {
        spq[n * DH + t] = pq [base + n * DH + t];
        spk[n * DH + t] = pkr[base + n * DH + t];
    }
    __syncthreads();
    float s = 0.f;
    for (int n = 0; n < NB; ++n) s += spk[n * DH + t];
    km[t] = s * (1.0f / NB);
    __syncthreads();
    for (int n = 0; n < NB; ++n) {
        float acc = 0.f;
        for (int d = 0; d < DH; ++d)
            acc += spq[t * DH + d] * (spk[n * DH + d] - km[d]);
        ssc[t * NB + n] = acc;
    }
    __syncthreads();
    for (int j = 0; j < TOPK; ++j) {
        float best = -3.0e38f;
        int   bi   = 0;
        for (int n = 0; n < NB; ++n) {
            float vv = ssc[t * NB + n];
            if (vv > best) { best = vv; bi = n; }
        }
        ssc[t * NB + bi] = -3.0e38f;
        lut[((size_t)bh * NB + t) * TOPK + j] = bi;
    }
}

// ---------------- Kernel 3: block-sparse flash attention ----------------
// grid = B*H*NB, 128 threads = 4 waves; wave w owns query rows [16w, 16w+16)
// Double-buffered LDS staging via GLOBAL_LOAD_ASYNC_TO_LDS_B128 (ASYNCcnt).
__global__ void __launch_bounds__(128)
sparse_attn_kernel(const _Float16* __restrict__ qh, const _Float16* __restrict__ kh,
                   const _Float16* __restrict__ vth, const int* __restrict__ lut,
                   float* __restrict__ out) {
    __shared__ _Float16 ldsK [2][BLK * LDSW];
    __shared__ _Float16 ldsVT[2][DH  * LDSW];
    __shared__ _Float16 ldsP [BLK * LDSW];

    int bid = blockIdx.x;
    int m   = bid & (NB - 1);
    int bh  = bid >> 6;
    int tid  = threadIdx.x;
    int wv   = tid >> 5;
    int lane = tid & 31;
    int hl   = lane >> 4;
    int li   = lane & 15;

    size_t base   = (size_t)bh * 4096 * DH;       // element base for q/k/out
    size_t vtbase = (size_t)bh * DH * 4096;       // element base for V^T

    // per-thread staging chunks: 512 chunks of 8 f16 per buffer
    int crow[4], ccol[4];
#pragma unroll
    for (int i = 0; i < 4; ++i) {
        int c = tid + i * 128;
        crow[i] = c >> 3;
        ccol[i] = (c & 7) * 8;
    }

    // Q A-fragments (loaded once, pre-scaled)
    const _Float16* qrow = qh + base + (size_t)(m * BLK + wv * 16 + li) * DH;
    h16v qa[2];
#pragma unroll
    for (int j = 0; j < 2; ++j)
        qa[j] = frag_a(qrow + 32 * j + 8 * hl);

    f8v O[4];
#pragma unroll
    for (int n = 0; n < 4; ++n)
#pragma unroll
        for (int v = 0; v < 8; ++v) O[n][v] = 0.f;

    float mrun[8], lrun[8];
#pragma unroll
    for (int v = 0; v < 8; ++v) { mrun[v] = -1.0e30f; lrun[v] = 0.f; }

    const int* lrow = lut + ((size_t)bh * NB + m) * TOPK;

    _Float16* bK = ldsK[0];  _Float16* nK = ldsK[1];
    _Float16* bV = ldsVT[0]; _Float16* nV = ldsVT[1];

    // prologue: stage block 0 (async), drain, publish
    {
        int kb = lrow[0];
        const _Float16* ksrc = kh  + base   + (size_t)kb * BLK * DH;
        const _Float16* vsrc = vth + vtbase + (size_t)kb * BLK;
#pragma unroll
        for (int i = 0; i < 4; ++i) {
            async_cp16(ksrc + crow[i] * DH + ccol[i],           &bK[crow[i] * LDSW + ccol[i]]);
            async_cp16(vsrc + (size_t)crow[i] * 4096 + ccol[i], &bV[crow[i] * LDSW + ccol[i]]);
        }
        async_drain();
    }
    __syncthreads();

#pragma unroll 1
    for (int t = 0; t < TOPK; ++t) {
        // kick off async staging of the next block into the shadow buffers;
        // the WMMA body below hides the latency
        bool hasNext = (t + 1 < TOPK);
        if (hasNext) {
            int kb2 = lrow[t + 1];
            const _Float16* ksrc = kh  + base   + (size_t)kb2 * BLK * DH;
            const _Float16* vsrc = vth + vtbase + (size_t)kb2 * BLK;
#pragma unroll
            for (int i = 0; i < 4; ++i) {
                async_cp16(ksrc + crow[i] * DH + ccol[i],           &nK[crow[i] * LDSW + ccol[i]]);
                async_cp16(vsrc + (size_t)crow[i] * 4096 + ccol[i], &nV[crow[i] * LDSW + ccol[i]]);
            }
        }

        // S strip = Q (16xD) * K^T (Dx64)
        f8v S[4];
#pragma unroll
        for (int n = 0; n < 4; ++n) {
            h16v kb0 = frag_b(&bK[(n * 16 + li) * LDSW + 8 * hl]);
            h16v kb1 = frag_b(&bK[(n * 16 + li) * LDSW + 32 + 8 * hl]);
            f8v z;
#pragma unroll
            for (int v = 0; v < 8; ++v) z[v] = 0.f;
            z    = wmma_f16(qa[0], kb0, z);
            S[n] = wmma_f16(qa[1], kb1, z);
        }

        // online softmax (DPP reductions, rows = VGPR idx + 8*half)
        float rs[8];
#pragma unroll
        for (int v = 0; v < 8; ++v) {
            float x = fmaxf(fmaxf(S[0][v], S[1][v]), fmaxf(S[2][v], S[3][v]));
            x = redmax16(x);
            float mn    = fmaxf(mrun[v], x);
            float alpha = __expf(mrun[v] - mn);
            mrun[v] = mn;
            lrun[v] *= alpha;
#pragma unroll
            for (int n = 0; n < 4; ++n) O[n][v] *= alpha;
            rs[v] = 0.f;
        }
#pragma unroll
        for (int n = 0; n < 4; ++n)
#pragma unroll
            for (int v = 0; v < 8; ++v) {
                float p = __expf(S[n][v] - mrun[v]);
                rs[v] += p;
                ldsP[(wv * 16 + v + 8 * hl) * LDSW + n * 16 + li] = (_Float16)p;
            }
#pragma unroll
        for (int v = 0; v < 8; ++v)
            lrun[v] += redsum16(rs[v]);

        // O += P (16x64) * V (64xD)
        h16v pa0 = frag_a(&ldsP[(wv * 16 + li) * LDSW + 8 * hl]);
        h16v pa1 = frag_a(&ldsP[(wv * 16 + li) * LDSW + 32 + 8 * hl]);
#pragma unroll
        for (int n = 0; n < 4; ++n) {
            h16v vb0 = frag_b(&bV[(n * 16 + li) * LDSW + 8 * hl]);
            h16v vb1 = frag_b(&bV[(n * 16 + li) * LDSW + 32 + 8 * hl]);
            O[n] = wmma_f16(pa0, vb0, O[n]);
            O[n] = wmma_f16(pa1, vb1, O[n]);
        }

        // drain async staging, publish, swap buffers
        if (hasNext) {
            async_drain();
            __syncthreads();
            _Float16* tmp;
            tmp = bK; bK = nK; nK = tmp;
            tmp = bV; bV = nV; nV = tmp;
        }
    }

    // epilogue: O /= l, write fp32 output
#pragma unroll
    for (int v = 0; v < 8; ++v) {
        float inv = 1.0f / lrun[v];
        int orow = m * BLK + wv * 16 + v + 8 * hl;
#pragma unroll
        for (int n = 0; n < 4; ++n)
            out[base + (size_t)orow * DH + n * 16 + li] = O[n][v] * inv;
    }
}

extern "C" void kernel_launch(void* const* d_in, const int* in_sizes, int n_in,
                              void* d_out, int out_size, void* d_ws, size_t ws_size,
                              hipStream_t stream) {
    (void)in_sizes; (void)n_in; (void)out_size; (void)ws_size;
    const float* q = (const float*)d_in[0];
    const float* k = (const float*)d_in[1];
    const float* v = (const float*)d_in[2];
    float* out = (float*)d_out;

    const size_t NE   = 2ull * 8 * 4096 * 64;   // elements per tensor
    const size_t BHNB = 2ull * 8 * NB;          // 1024

    char* ws = (char*)d_ws;
    _Float16* qh  = (_Float16*)(ws);
    _Float16* kh  = (_Float16*)(ws + NE * 2);
    _Float16* vt  = (_Float16*)(ws + NE * 4);
    float*    pq  = (float*)   (ws + NE * 6);
    float*    pkr = (float*)   (ws + NE * 6 + BHNB * DH * 4);
    int*      lut = (int*)     (ws + NE * 6 + BHNB * DH * 8);

    cvt_kernel<<<(int)((NE + 255) / 256), 256, 0, stream>>>(q, k, qh, kh, (int)NE);
    vtrans_kernel<<<(int)BHNB, 256, 0, stream>>>(v, vt);
    pool_kernel<<<(int)BHNB, 64, 0, stream>>>(q, k, pq, pkr);
    topk_kernel<<<16, 64, 0, stream>>>(pq, pkr, lut);
    sparse_attn_kernel<<<(int)BHNB, 128, 0, stream>>>(qh, kh, vt, lut, out);
}
